// HungarianMatcher_22883585753393
// MI455X (gfx1250) — matrix operand order repaired and checked by reading, not compiled, not run on hardware
//
#include <hip/hip_runtime.h>
#include <stdint.h>

#define NQ 400
#define NM 200000
#define NG 64
#define NC 20

typedef __attribute__((ext_vector_type(16))) _Float16 v16h;
typedef __attribute__((ext_vector_type(8)))  float    v8f;
typedef __attribute__((ext_vector_type(4)))  float    v4f;
typedef __attribute__((ext_vector_type(4)))  unsigned v4u;
typedef __attribute__((ext_vector_type(2)))  unsigned v2u;

#define LSTRIDE 20   // LDS row stride in uints: 80B = 5*16B -> rows stay 16B
                     // aligned for b128 LDS ops; gcd(20,64)=4 -> 16 rows hit
                     // 16 distinct banks (conflict-free fragment reads)

__device__ __forceinline__ unsigned pack2(float a, float b) {
    // single v_cvt_pk_rtz_f16_f32 instead of cvt+cvt+shift+or
    return __builtin_bit_cast(unsigned, __builtin_amdgcn_cvt_pkrtz(a, b));
}

__device__ __forceinline__ float sigmoidf(float x) {
    return __builtin_amdgcn_rcpf(1.f + __expf(-x));
}

// ---------------- kernel 1: per-instance mask sums (deterministic tree) ----
__global__ __launch_bounds__(256) void hm_mask_sum(const float* __restrict__ inst,
                                                   float* __restrict__ maskSum) {
    __shared__ float s[256];
    const int g = blockIdx.x;
    const int t = threadIdx.x;
    const float* row = inst + (size_t)g * NM;
    float acc = 0.f;
    for (int k = t * 4; k < NM; k += 256 * 4) {   // NM % 4 == 0
        v4f x = *(const v4f*)(row + k);
        acc += (x[0] + x[1]) + (x[2] + x[3]);
    }
    s[t] = acc;
    __syncthreads();
    for (int off = 128; off > 0; off >>= 1) {
        if (t < off) s[t] += s[t + off];
        __syncthreads();
    }
    if (t == 0) maskSum[g] = s[0];
}

// ---------------- kernel 2: sigmoid + WMMA GEMM + prob row-sums ------------
// grid = (NQ/16, ksplit); block = 128 (4 waves). Wave w owns instances
// [16w, 16w+16). Double-buffered LDS pipeline: unconditional b128 global
// loads for tile i+1 are issued before tile i's v_wmma_f32_16x16x32_f16,
// packed (v_cvt_pk_rtz_f16_f32) and stored via ds_store_b128 after, one
// barrier per K-step.
__global__ __launch_bounds__(128) void hm_gemm(const float* __restrict__ mask_logit,
                                               const float* __restrict__ inst_masks,
                                               float* __restrict__ wsNum,
                                               float* __restrict__ wsRow,
                                               int kchunk) {
    __shared__ __align__(16) unsigned aS[2][16][LSTRIDE]; // probs tile (f16 pairs)
    __shared__ __align__(16) unsigned bS[2][64][LSTRIDE]; // masks tile (f16 pairs)
    __shared__ float pS[128];                             // rowsum partials

    const int qt   = blockIdx.x;
    const int kb   = blockIdx.y;
    const int tid  = threadIdx.x;
    const int lane = tid & 31;
    const int wave = tid >> 5;

    const int q0      = qt * 16;
    const int k_begin = kb * kchunk;
    int k_end = k_begin + kchunk;
    if (k_end > NM) k_end = NM;

    // cooperative staging assignments
    const int ra = tid >> 3;          // A row 0..15  (8 threads/row, 4 floats each)
    const int ca = (tid & 7) * 4;     // A col 0..28
    const int rb = tid >> 1;          // B row 0..63  (2 threads/row, 16 floats each)
    const int cb = (tid & 1) * 16;    // B col 0 or 16

    const float* aBase = mask_logit + (size_t)(q0 + ra) * NM;
    const float* bBase = inst_masks + (size_t)rb * NM;

    // fragment mapping constants (CDNA5 16-bit 16x32 A / 32x16 B layouts)
    const int arow = lane & 15;
    const int aoff = (lane >> 4) * 4;          // lanes 16-31 hold K+8 halves
    const int brow = (lane & 15) + wave * 16;  // instance (N) this lane holds
    const int boff = (lane >> 4) * 4;          // lanes 16-31 hold K+8 halves

    v8f acc = {};
    float rowAcc = 0.f;

    const int nFull = (k_end - k_begin) >> 5;  // guard-free 32-wide steps
    const int kTail = k_begin + (nFull << 5);

    // ---- staging helpers --------------------------------------------------
    auto stage_load = [&](int k0, v4f& a4, v4f b4[4]) {
        // A streams once over 320MB: non-temporal so L2 keeps inst_masks
        a4 = __builtin_nontemporal_load((const v4f*)(aBase + k0 + ca));
#pragma unroll
        for (int j = 0; j < 4; ++j)
            b4[j] = *(const v4f*)(bBase + k0 + cb + 4 * j);
    };
    auto stage_store = [&](int buf, const v4f& a4, const v4f b4[4]) {
        float p0 = sigmoidf(a4[0]), p1 = sigmoidf(a4[1]);
        float p2 = sigmoidf(a4[2]), p3 = sigmoidf(a4[3]);
        rowAcc += (p0 + p1) + (p2 + p3);
        v2u aw = { pack2(p0, p1), pack2(p2, p3) };
        *(v2u*)&aS[buf][ra][ca >> 1] = aw;                 // ds_store_b64
        v4u bw0 = { pack2(b4[0][0], b4[0][1]), pack2(b4[0][2], b4[0][3]),
                    pack2(b4[1][0], b4[1][1]), pack2(b4[1][2], b4[1][3]) };
        v4u bw1 = { pack2(b4[2][0], b4[2][1]), pack2(b4[2][2], b4[2][3]),
                    pack2(b4[3][0], b4[3][1]), pack2(b4[3][2], b4[3][3]) };
        *(v4u*)&bS[buf][rb][(cb >> 1) + 0] = bw0;          // ds_store_b128
        *(v4u*)&bS[buf][rb][(cb >> 1) + 4] = bw1;          // ds_store_b128
    };
    auto mma_step = [&](int buf) {
        union { v4u q[2]; v16h h; } a, b;
        a.q[0] = *(const v4u*)&aS[buf][arow][aoff];        // ds_load_b128: K 0..7 (or 8..15)
        a.q[1] = *(const v4u*)&aS[buf][arow][8 + aoff];    // ds_load_b128: K 16..23 (or 24..31)
        b.q[0] = *(const v4u*)&bS[buf][brow][boff];        // ds_load_b128: K 0..7  (or 8..15)
        b.q[1] = *(const v4u*)&bS[buf][brow][boff + 8];    // ds_load_b128: K 16..23 (or 24..31)
        acc = __builtin_amdgcn_wmma_f32_16x16x32_f16(
                  false, a.h, false, b.h, (short)0, acc, false, false);
    };

    // ---- pipelined main loop (no bounds checks) ---------------------------
    int buf = 0;
    if (nFull > 0) {
        v4f a4; v4f b4[4];
        stage_load(k_begin, a4, b4);
        stage_store(0, a4, b4);
        __syncthreads();
        for (int it = 0; it < nFull; ++it) {
            const int k0 = k_begin + (it << 5);
            v4f na; v4f nb[4];
            const bool hasNext = (it + 1 < nFull);
            if (hasNext) stage_load(k0 + 32, na, nb);
            if (it + 2 < nFull) {       // global_prefetch_b8, 2 tiles ahead
                __builtin_prefetch(aBase + k0 + 64 + ca, 0, 3);
                __builtin_prefetch(bBase + k0 + 64 + cb, 0, 3);
            }
            mma_step(buf);
            if (hasNext) stage_store(buf ^ 1, na, nb);
            __syncthreads();
            buf ^= 1;
        }
    }

    // ---- tail step (at most one, bounds-guarded; absent for default split)
    if (kTail < k_end) {
        {
            float p[4];
#pragma unroll
            for (int i = 0; i < 4; ++i) {
                const int k = kTail + ca + i;
                float x  = (k < k_end) ? aBase[k] : 0.f;
                float pp = (k < k_end) ? sigmoidf(x) : 0.f;
                p[i] = pp;
                rowAcc += pp;
            }
            aS[buf][ra][(ca >> 1) + 0] = pack2(p[0], p[1]);
            aS[buf][ra][(ca >> 1) + 1] = pack2(p[2], p[3]);
#pragma unroll
            for (int j = 0; j < 8; ++j) {
                const int k = kTail + cb + 2 * j;
                float x0 = (k     < k_end) ? bBase[k]     : 0.f;
                float x1 = (k + 1 < k_end) ? bBase[k + 1] : 0.f;
                bS[buf][rb][(cb >> 1) + j] = pack2(x0, x1);
            }
        }
        __syncthreads();
        mma_step(buf);
        __syncthreads();
    }

    // ---- write 16x16 partial tile (C/D layout: VGPR r, lanes 16-31 -> M+8)
#pragma unroll
    for (int r = 0; r < 8; ++r) {
        const int m = r + 8 * (lane >> 4);
        const int n = (lane & 15) + wave * 16;
        wsNum[((size_t)kb * NQ + (q0 + m)) * NG + n] = acc[r];
    }

    // ---- deterministic rowsum partial (8 threads per row, fixed order)
    pS[tid] = rowAcc;
    __syncthreads();
    if (tid < 16) {
        float s = 0.f;
#pragma unroll
        for (int i = 0; i < 8; ++i) s += pS[tid * 8 + i];
        wsRow[(size_t)kb * NQ + (q0 + tid)] = s;
    }
}

// ---------------- kernel 3: reduce partials + softmax class cost + dice ----
__global__ __launch_bounds__(64) void hm_final(const float* __restrict__ sem_logit,
                                               const int* __restrict__ sem_labels,
                                               const float* __restrict__ wsNum,
                                               const float* __restrict__ wsRow,
                                               const float* __restrict__ maskSum,
                                               float* __restrict__ out, int ksplit) {
    const int q = blockIdx.x;
    const int g = threadIdx.x;

    float num = 0.f;
    for (int kb = 0; kb < ksplit; ++kb)
        num += wsNum[((size_t)kb * NQ + q) * NG + g];
    float rs = 0.f;
    for (int kb = 0; kb < ksplit; ++kb)
        rs += wsRow[(size_t)kb * NQ + q];

    const float* sl = sem_logit + q * NC;
    float mx = sl[0];
#pragma unroll
    for (int c = 1; c < NC; ++c) mx = fmaxf(mx, sl[c]);
    float den = 0.f;
#pragma unroll
    for (int c = 0; c < NC; ++c) den += __expf(sl[c] - mx);
    const int lbl = sem_labels[g];
    const float cls = -__expf(sl[lbl] - mx) / den;

    const float dice = 1.f - (2.f * num + 1.f) / (rs + maskSum[g] + 1.f);
    out[q * NG + g] = cls + dice;
}

extern "C" void kernel_launch(void* const* d_in, const int* in_sizes, int n_in,
                              void* d_out, int out_size, void* d_ws, size_t ws_size,
                              hipStream_t stream) {
    const float* mask_logit = (const float*)d_in[0];
    const float* sem_logit  = (const float*)d_in[1];
    const float* inst_masks = (const float*)d_in[2];
    const int*   sem_labels = (const int*)d_in[3];
    float* out = (float*)d_out;
    (void)in_sizes; (void)n_in; (void)out_size;

    // K-split sized to workspace (deterministic: ws_size constant per run).
    long cap = (long)(ws_size / 4) - NG;           // floats available for partials
    long per = (long)NQ * (NG + 1);                // floats per K-split slice
    int ksplit = 25;                               // 25 -> kchunk = 8000 = 250*32
    if (cap < (long)ksplit * per) {
        ksplit = (int)(cap / per);
        if (ksplit < 1) ksplit = 1;
    }
    const int kchunk = (NM + ksplit - 1) / ksplit;

    float* wsNum   = (float*)d_ws;
    float* wsRow   = wsNum + (size_t)ksplit * NQ * NG;
    float* maskSum = wsRow + (size_t)ksplit * NQ;

    hm_mask_sum<<<NG, 256, 0, stream>>>(inst_masks, maskSum);
    hm_gemm<<<dim3(NQ / 16, ksplit), 128, 0, stream>>>(mask_logit, inst_masks,
                                                       wsNum, wsRow, kchunk);
    hm_final<<<NQ, NG, 0, stream>>>(sem_logit, sem_labels, wsNum, wsRow,
                                    maskSum, out, ksplit);
}
